// ReadoutLayer_11381663334472
// MI455X (gfx1250) — matrix operand order repaired and checked by reading, not compiled, not run on hardware
//
#include <hip/hip_runtime.h>

#define DIN   1024
#define DOUT  128
#define TT    500
#define BB    64
#define SCHUNK 5
#define NCHUNKS (TT / SCHUNK)      // 100
#define LDS_ROW (DIN + 8)          // ushort elems; +16B/row pad to spread LDS banks

typedef __attribute__((ext_vector_type(16))) __bf16       v16bf;
typedef __attribute__((ext_vector_type(2)))  __bf16       v2bf;
typedef __attribute__((ext_vector_type(8)))  float        v8f;
typedef __attribute__((ext_vector_type(4)))  unsigned int u32x4;

union Frag16 { v16bf v; u32x4 q[2]; };

// pack two f32 -> two bf16 in one dword (lo -> [15:0], hi -> [31:16])
__device__ __forceinline__ unsigned int pk_bf16(float lo, float hi) {
#if __has_builtin(__builtin_amdgcn_cvt_pk_bf16_f32)
  auto t = __builtin_amdgcn_cvt_pk_bf16_f32(lo, hi);
  unsigned int u;
  __builtin_memcpy(&u, &t, 4);
  return u;
#else
  union { v2bf b; unsigned int u; } cv;
  cv.b[0] = (__bf16)lo;            // fptrunc RNE; backend selects v_cvt_pk_bf16_f32
  cv.b[1] = (__bf16)hi;
  return cv.u;
#endif
}

__device__ __forceinline__ unsigned short f2bf(float f) {
  union { __bf16 b[2]; unsigned int u; } cv;
  cv.b[0] = (__bf16)f;
  cv.b[1] = (__bf16)0.0f;
  return (unsigned short)(cv.u & 0xFFFFu);
}

// ---------------------------------------------------------------------------
// Prep: wT[d][c] = bf16(w[c][d]);  scale[d] = 1/(sum_c w^2 + eps);
//       coeff[s][d] = (1 - beta[d]^(T-s)) / T   (backward recurrence, stable)
// ---------------------------------------------------------------------------
__global__ __launch_bounds__(256) void rl_prep(
    const float* __restrict__ w, const float* __restrict__ beta,
    float* __restrict__ coeff, float* __restrict__ scale,
    unsigned short* __restrict__ wT)
{
  int tid = blockIdx.x * blockDim.x + threadIdx.x;
  if (tid < DIN * DOUT) {
    int c = tid & (DIN - 1);
    int d = tid >> 10;
    wT[tid] = f2bf(w[(size_t)c * DOUT + d]);
  }
  if (tid < DOUT) {
    float nrm = 0.f;
    for (int c = 0; c < DIN; ++c) {
      float v = w[(size_t)c * DOUT + tid];
      nrm = fmaf(v, v, nrm);
    }
    scale[tid] = 1.0f / (nrm + 1e-8f);
    float bt = beta[tid];
    float pw = 1.0f;
    const float invT = 1.0f / (float)TT;
    for (int s = TT - 1; s >= 0; --s) {
      pw *= bt;                               // pw = beta^(T-s)
      coeff[s * DOUT + tid] = (1.0f - pw) * invT;
    }
  }
}

// ---------------------------------------------------------------------------
// Main: per block = (b-tile of 16, chunk of 5 time steps). 8 waves, one
// 16-wide N-tile each. Per step: stage x tile as bf16 in LDS, 32 bf16 WMMAs
// per wave (K=1024), scale h-tile by coeff[s][d], accumulate in f32.
// Partials written to disjoint slabs (deterministic reduction later).
// ---------------------------------------------------------------------------
__global__ __launch_bounds__(256) void rl_main(
    const float* __restrict__ x, const unsigned short* __restrict__ wT,
    const float* __restrict__ coeff, float* __restrict__ partial)
{
  __shared__ unsigned short Abuf[16 * LDS_ROW];   // 33 KB

  const int btile = blockIdx.x & 3;
  const int chunk = blockIdx.x >> 2;
  const int s0   = chunk * SCHUNK;
  const int sEnd = s0 + SCHUNK;

  const int lane = threadIdx.x & 31;
  const int wave = threadIdx.x >> 5;
  const int col  = lane & 15;        // N index / A row (M)
  const int half = lane >> 4;        // K-half selector
  const int dbase = wave * 16;

  // cooperative x loader mapping: 16 rows x 16 segs of 64 floats
  const int r   = threadIdx.x >> 4;
  const int seg = threadIdx.x & 15;
  const float* src0 = x + (size_t)(btile * 16 + r) * TT * DIN + (size_t)seg * 64;

  v8f acc = {0.f, 0.f, 0.f, 0.f, 0.f, 0.f, 0.f, 0.f};

  const unsigned short* arow = &Abuf[col * LDS_ROW + half * 8];
  const unsigned short* brow = wT + (size_t)(dbase + col) * DIN + half * 8;

  for (int s = s0; s < sEnd; ++s) {
    // ---- stage A tile (16 x 1024 f32 -> bf16 -> LDS) ----
    const float4* s4 = reinterpret_cast<const float4*>(src0 + (size_t)s * DIN);
    u32x4* d4 = reinterpret_cast<u32x4*>(&Abuf[r * LDS_ROW + seg * 64]);
#pragma unroll
    for (int j = 0; j < 8; ++j) {
      float4 f0 = s4[2 * j];
      float4 f1 = s4[2 * j + 1];
      u32x4 p;
      p.x = pk_bf16(f0.x, f0.y);
      p.y = pk_bf16(f0.z, f0.w);
      p.z = pk_bf16(f1.x, f1.y);
      p.w = pk_bf16(f1.z, f1.w);
      d4[j] = p;
    }
    if (s + 1 < sEnd)
      __builtin_prefetch(src0 + (size_t)(s + 1) * DIN, 0, 0);
    __syncthreads();

    // ---- h = x_tile @ w_tile over K=1024 ----
    v8f h = {0.f, 0.f, 0.f, 0.f, 0.f, 0.f, 0.f, 0.f};
#pragma unroll 4
    for (int kb = 0; kb < DIN; kb += 32) {
      Frag16 a, b;
      a.q[0] = *reinterpret_cast<const u32x4*>(arow + kb);       // K kb+half*8 ..+7
      a.q[1] = *reinterpret_cast<const u32x4*>(arow + kb + 16);  // K kb+16+half*8 ..+7
      const u32x4* bp = reinterpret_cast<const u32x4*>(brow + kb);
      b.q[0] = bp[0];                                            // K kb+half*8 ..+7
      b.q[1] = bp[2];                                            // K kb+16+half*8 ..+7
      h = __builtin_amdgcn_wmma_f32_16x16x32_bf16(
              false, a.v, false, b.v, (short)0, h, false, false);
    }
    __syncthreads();

    // ---- weight by coeff[s][d] (per-lane scalar: d = dbase + N) ----
    const float cf = coeff[s * DOUT + dbase + col];
#pragma unroll
    for (int i = 0; i < 8; ++i) acc[i] = fmaf(h[i], cf, acc[i]);
  }

  // disjoint partial slab: [chunk][64][128]
  float* pout = partial + (size_t)chunk * (BB * DOUT)
                        + (size_t)(btile * 16) * DOUT + dbase + col;
#pragma unroll
  for (int i = 0; i < 8; ++i)
    pout[(i + half * 8) * DOUT] = acc[i];      // M = i + half*8, N = col
}

// ---------------------------------------------------------------------------
// Final: fixed-order reduction over chunks, apply 1/(norm+eps), subtract b
// ---------------------------------------------------------------------------
__global__ __launch_bounds__(256) void rl_fin(
    const float* __restrict__ partial, const float* __restrict__ scale,
    const float* __restrict__ bvec, float* __restrict__ out)
{
  int tid = blockIdx.x * blockDim.x + threadIdx.x;
  if (tid >= BB * DOUT) return;
  int d = tid & (DOUT - 1);
  float s = 0.f;
  for (int ch = 0; ch < NCHUNKS; ++ch)
    s += partial[(size_t)ch * (BB * DOUT) + tid];
  out[tid] = s * scale[d] - bvec[d];
}

extern "C" void kernel_launch(void* const* d_in, const int* in_sizes, int n_in,
                              void* d_out, int out_size, void* d_ws, size_t ws_size,
                              hipStream_t stream) {
  const float* x    = (const float*)d_in[0];   // [64,500,1024]
  const float* w    = (const float*)d_in[1];   // [1024,128]
  const float* beta = (const float*)d_in[2];   // [128]
  const float* bvec = (const float*)d_in[3];   // [128]
  float* out = (float*)d_out;                  // [64,128]

  // workspace layout (float units; all 16B aligned)
  float* ws = (float*)d_ws;
  float* coeff = ws;                                   // 500*128 = 64000
  float* scale = ws + TT * DOUT;                       // 128
  unsigned short* wT = (unsigned short*)(ws + TT * DOUT + DOUT);       // 1024*128 u16
  float* partial = ws + TT * DOUT + DOUT + (DIN * DOUT / 2);           // 100*64*128

  rl_prep<<<(DIN * DOUT + 255) / 256, 256, 0, stream>>>(w, beta, coeff, scale, wT);
  rl_main<<<4 * NCHUNKS, 256, 0, stream>>>(x, wT, coeff, partial);
  rl_fin<<<(BB * DOUT + 255) / 256, 256, 0, stream>>>(partial, scale, bvec, out);
}
